// Model_37211596653320
// MI455X (gfx1250) — matrix operand order repaired and checked by reading, not compiled, not run on hardware
//
#include <hip/hip_runtime.h>
#include <hip/hip_bf16.h>
#include <math.h>

typedef __attribute__((ext_vector_type(16))) __bf16 v16bf;
typedef __attribute__((ext_vector_type(2)))  __bf16 v2bf;
typedef __attribute__((ext_vector_type(8)))  float  v8f;

// ---------------------------------------------------------------------------
// f32 -> bf16 conversion (grid-stride)
// ---------------------------------------------------------------------------
__global__ void f32_to_bf16_kernel(const float* __restrict__ in,
                                   __bf16* __restrict__ out, int n) {
    for (int i = blockIdx.x * blockDim.x + threadIdx.x; i < n;
         i += gridDim.x * blockDim.x) {
        out[i] = (__bf16)in[i];
    }
}

// ---------------------------------------------------------------------------
// Implicit-GEMM 3x3 SAME conv + bias + ReLU via WMMA bf16 -> f32.
//   A = weights  [Cout][K]  (K = Cin*9, OIHW flat)
//   B = im2col   [K][Ngemm] (Ngemm = Nimg*Hs*Hs)
// Block: 256 threads = 8 waves arranged 2(M) x 4(N). Block tile 64 x 128.
// Wave tile 32 x 32 = 4 WMMA accumulators per K=32 step.
// LDS tiles are stored PRE-SWIZZLED in WMMA fragment layout:
//   frag[panel][lane][16 bf16]  -> each lane's operand is one contiguous
//   32-byte LDS read (2 x ds_load_b128) instead of 16+ ds_load_u16.
// Fragment element mapping (wave32, 16-bit A 16x32 / B 32x16, ISA 7.12.2):
//   A: lane = 16*h + m,   elem e=2i+t  <->  k = (i<4 ? 2i : 16+2(i-4)) + 8h + t
//   B: lane = 16*h + col, elem e=2i+t  <->  k = (i<4 ? 2i : 16+2(i-4)) + 8h + t
// ---------------------------------------------------------------------------
__global__ __launch_bounds__(256) void conv_gemm_wmma(
    const __bf16* __restrict__ in,    // (Nimg, Cin, Hs, Hs)
    const __bf16* __restrict__ wgt,   // (Cout, K)
    const float*  __restrict__ bias,  // (Cout)
    __bf16* __restrict__ out,         // (Nimg, Cout, Hs, Hs)
    int Nimg, int Cin, int Hs, int Cout)
{
    const int K     = Cin * 9;
    const int HW    = Hs * Hs;
    const int Ngemm = Nimg * HW;

    const int tid  = threadIdx.x;
    const int lane = tid & 31;
    const int wave = tid >> 5;
    const int h16  = lane >> 4;
    const int col  = lane & 15;

    const int m_base = blockIdx.y * 64;
    const int n_base = blockIdx.x * 128;
    const int wm = wave >> 2;   // 0..1 : M offset 32*wm
    const int wn = wave & 3;    // 0..3 : N offset 32*wn

    __shared__ alignas(32) __bf16 As[4 * 512];   // 4 panels of 16 rows
    __shared__ alignas(32) __bf16 Bs[8 * 512];   // 8 panels of 16 cols

    v8f acc00 = {}, acc01 = {}, acc10 = {}, acc11 = {};

    for (int kt = 0; kt < K; kt += 32) {
        // ---- A fill: 1024 packed bf16 pairs, fragment-layout stores ----
        for (int idx = tid; idx < 1024; idx += 256) {
            int i = idx & 7, ln = (idx >> 3) & 31, panel = idx >> 8;
            int m = ln & 15, h = ln >> 4;
            int k0 = ((i < 4) ? (2 * i) : (16 + 2 * (i - 4))) + 8 * h;
            int gm = m_base + panel * 16 + m;
            int gk = kt + k0;
            v2bf pr;
            pr[0] = (__bf16)0.0f;
            pr[1] = (__bf16)0.0f;
            if (gm < Cout) {
                const __bf16* wrow = wgt + (size_t)gm * K;
                if (gk < K)     pr[0] = wrow[gk];
                if (gk + 1 < K) pr[1] = wrow[gk + 1];
            }
            *(v2bf*)&As[panel * 512 + ln * 16 + 2 * i] = pr;
        }
        // ---- B fill (im2col gather): 2048 packed pairs ----
        for (int idx = tid; idx < 2048; idx += 256) {
            int ep = idx & 7, ln = (idx >> 3) & 31, pn = idx >> 8;
            int c = ln & 15, h = ln >> 4;
            int kk = ((ep < 4) ? (2 * ep) : (16 + 2 * (ep - 4))) + 8 * h;
            int gn = n_base + pn * 16 + c;
            v2bf pr;
            pr[0] = (__bf16)0.0f;
            pr[1] = (__bf16)0.0f;
            if (gn < Ngemm) {
                int img = gn / HW, rem = gn - img * HW;
                int y0 = rem / Hs, x0 = rem - y0 * Hs;
                #pragma unroll
                for (int t = 0; t < 2; ++t) {
                    int gk = kt + kk + t;
                    if (gk < K) {
                        int ci = gk / 9, rs = gk - 9 * ci;
                        int rr = rs / 3;
                        int y = y0 + rr - 1, x = x0 + (rs - 3 * rr) - 1;
                        if ((unsigned)y < (unsigned)Hs && (unsigned)x < (unsigned)Hs)
                            pr[t] = in[((size_t)(img * Cin + ci) * Hs + y) * Hs + x];
                    }
                }
            }
            *(v2bf*)&Bs[pn * 512 + ln * 16 + 2 * ep] = pr;
        }
        __syncthreads();

        // gfx1250 prefetch of next weight K-tile (global_prefetch_b8)
        if (kt + 32 < K) {
            int pm = m_base + (tid & 63);
            if (pm < Cout)
                __builtin_prefetch((const void*)&wgt[(size_t)pm * K + kt + 32], 0, 1);
        }

        // ---- fragments: contiguous 32B LDS reads (2x ds_load_b128 each) ----
        v16bf a0 = *(const v16bf*)&As[(2 * wm + 0) * 512 + lane * 16];
        v16bf a1 = *(const v16bf*)&As[(2 * wm + 1) * 512 + lane * 16];
        v16bf b0 = *(const v16bf*)&Bs[(2 * wn + 0) * 512 + lane * 16];
        v16bf b1 = *(const v16bf*)&Bs[(2 * wn + 1) * 512 + lane * 16];

        acc00 = __builtin_amdgcn_wmma_f32_16x16x32_bf16(
                    false, a0, false, b0, (short)0, acc00, false, false);
        acc01 = __builtin_amdgcn_wmma_f32_16x16x32_bf16(
                    false, a0, false, b1, (short)0, acc01, false, false);
        acc10 = __builtin_amdgcn_wmma_f32_16x16x32_bf16(
                    false, a1, false, b0, (short)0, acc10, false, false);
        acc11 = __builtin_amdgcn_wmma_f32_16x16x32_bf16(
                    false, a1, false, b1, (short)0, acc11, false, false);
        __syncthreads();
    }

    // ---- epilogue: bias + ReLU, scatter to NCHW bf16 ----
    // C/D layout: VGPR i -> M = i + 8*h16, N = col
    #pragma unroll
    for (int mi = 0; mi < 2; ++mi) {
        #pragma unroll
        for (int i = 0; i < 8; ++i) {
            int gm = m_base + (2 * wm + mi) * 16 + i + 8 * h16;
            if (gm >= Cout) continue;
            float bv = bias[gm];
            #pragma unroll
            for (int ni = 0; ni < 2; ++ni) {
                int gn = n_base + (2 * wn + ni) * 16 + col;
                if (gn < Ngemm) {
                    float v = bv;
                    if (mi == 0) v += (ni == 0) ? acc00[i] : acc01[i];
                    else         v += (ni == 0) ? acc10[i] : acc11[i];
                    v = v > 0.0f ? v : 0.0f;
                    int img = gn / HW, rem = gn - img * HW;
                    out[(size_t)(img * Cout + gm) * HW + rem] = (__bf16)v;
                }
            }
        }
    }
}

// ---------------------------------------------------------------------------
// 2x2 max pool, stride 2, VALID (bf16 -> bf16)
// ---------------------------------------------------------------------------
__global__ void maxpool2_kernel(const __bf16* __restrict__ in,
                                __bf16* __restrict__ out, int NC, int Hs) {
    int Hout = Hs >> 1;
    int total = NC * Hout * Hout;
    for (int idx = blockIdx.x * blockDim.x + threadIdx.x; idx < total;
         idx += gridDim.x * blockDim.x) {
        int p  = idx % (Hout * Hout);
        int nc = idx / (Hout * Hout);
        int py = p / Hout, px = p % Hout;
        const __bf16* base = in + (size_t)nc * Hs * Hs;
        int y0 = 2 * py, x0 = 2 * px;
        float a = (float)base[y0 * Hs + x0];
        float b = (float)base[y0 * Hs + x0 + 1];
        float c = (float)base[(y0 + 1) * Hs + x0];
        float d = (float)base[(y0 + 1) * Hs + x0 + 1];
        float m = fmaxf(fmaxf(a, b), fmaxf(c, d));
        out[idx] = (__bf16)m;
    }
}

// ---------------------------------------------------------------------------
// Prototypes: ftr_avg[b,k,c] = l2norm_c( mean_hw( sum_n y[b,n,k]*f / sum y ) )
// One block per (b,k). f = final features (400, 512, 5, 5) bf16.
// ---------------------------------------------------------------------------
__global__ __launch_bounds__(256) void proto_kernel(
    const __bf16* __restrict__ f, const float* __restrict__ ytrain,
    float* __restrict__ proto)  // (4,5,512)
{
    int b = blockIdx.x / 5, k = blockIdx.x % 5;
    __shared__ float sm[512];
    __shared__ float red[256];
    int tid = threadIdx.x;

    for (int c = tid; c < 512; c += 256) {
        float acc = 0.0f, wsum = 0.0f;
        for (int n = 0; n < 25; ++n) {
            float wv = ytrain[(b * 25 + n) * 5 + k];
            wsum += wv;
            if (wv != 0.0f) {
                const __bf16* fp = f + ((size_t)(b * 25 + n) * 512 + c) * 25;
                float s = 0.0f;
                #pragma unroll
                for (int p = 0; p < 25; ++p) s += (float)fp[p];
                acc += wv * s;
            }
        }
        sm[c] = acc / (25.0f * fmaxf(wsum, 1e-12f));
    }
    __syncthreads();

    float ss = 0.0f;
    for (int c = tid; c < 512; c += 256) ss += sm[c] * sm[c];
    red[tid] = ss;
    __syncthreads();
    for (int s = 128; s > 0; s >>= 1) {
        if (tid < s) red[tid] += red[tid + s];
        __syncthreads();
    }
    float scale = 1.0f / fmaxf(sqrtf(red[0]), 1e-12f);
    for (int c = tid; c < 512; c += 256)
        proto[(size_t)(b * 5 + k) * 512 + c] = sm[c] * scale;
}

// ---------------------------------------------------------------------------
// Classifier head: predict[q, cls] = l2norm(GAP(f_query)) . wc[cls] + bc[cls]
// One block per query (300 blocks).
// ---------------------------------------------------------------------------
__global__ __launch_bounds__(256) void classifier_kernel(
    const __bf16* __restrict__ f, const float* __restrict__ wc,
    const float* __restrict__ bc, float* __restrict__ outp)  // (300,64)
{
    int q = blockIdx.x;          // b*75 + j
    int img = 100 + q;
    __shared__ float sq[512];
    __shared__ float red[256];
    int tid = threadIdx.x;

    for (int c = tid; c < 512; c += 256) {
        const __bf16* fp = f + ((size_t)img * 512 + c) * 25;
        float s = 0.0f;
        #pragma unroll
        for (int p = 0; p < 25; ++p) s += (float)fp[p];
        sq[c] = s * (1.0f / 25.0f);
    }
    __syncthreads();

    float ss = 0.0f;
    for (int c = tid; c < 512; c += 256) ss += sq[c] * sq[c];
    red[tid] = ss;
    __syncthreads();
    for (int s = 128; s > 0; s >>= 1) {
        if (tid < s) red[tid] += red[tid + s];
        __syncthreads();
    }
    float scale = 1.0f / fmaxf(sqrtf(red[0]), 1e-12f);

    if (tid < 64) {
        const float* w = wc + (size_t)tid * 512;
        float d = bc[tid];
        for (int c = 0; c < 512; ++c) d += sq[c] * scale * w[c];
        outp[(size_t)q * 64 + tid] = d;
    }
}

// ---------------------------------------------------------------------------
// Cosine scores: one wave32 per (query, spatial position). 7500 waves.
//   score[q, i, p] = (f[q,:,p] . proto[b,i,:]) / max(||f[q,:,p]||, eps)
// ---------------------------------------------------------------------------
__global__ __launch_bounds__(256) void cosine_kernel(
    const __bf16* __restrict__ f, const float* __restrict__ proto,
    float* __restrict__ outs)  // (300, 5, 25)
{
    int gw = blockIdx.x * 8 + (threadIdx.x >> 5);
    if (gw >= 300 * 25) return;
    int lane = threadIdx.x & 31;
    int q = gw / 25, p = gw % 25;
    int b = q / 75;
    int img = 100 + q;

    float dots[5] = {0.f, 0.f, 0.f, 0.f, 0.f};
    float ss = 0.f;
    for (int c = lane; c < 512; c += 32) {
        float v = (float)f[((size_t)img * 512 + c) * 25 + p];
        ss += v * v;
        #pragma unroll
        for (int i = 0; i < 5; ++i)
            dots[i] += v * proto[(size_t)(b * 5 + i) * 512 + c];
    }
    #pragma unroll
    for (int off = 16; off > 0; off >>= 1) {
        ss += __shfl_xor(ss, off);
        #pragma unroll
        for (int i = 0; i < 5; ++i) dots[i] += __shfl_xor(dots[i], off);
    }
    if (lane == 0) {
        float scale = 1.0f / fmaxf(sqrtf(ss), 1e-12f);
        #pragma unroll
        for (int i = 0; i < 5; ++i)
            outs[((size_t)q * 5 + i) * 25 + p] = dots[i] * scale;
    }
}

// ---------------------------------------------------------------------------
// Host-side launch
// ---------------------------------------------------------------------------
static inline size_t align256(size_t x) { return (x + 255) & ~(size_t)255; }
static inline int    cdiv(int a, int b) { return (a + b - 1) / b; }

extern "C" void kernel_launch(void* const* d_in, const int* in_sizes, int n_in,
                              void* d_out, int out_size, void* d_ws, size_t ws_size,
                              hipStream_t stream) {
    (void)in_sizes; (void)n_in; (void)out_size; (void)ws_size;

    const float* xtrain = (const float*)d_in[0];   // (4,25,3,84,84)
    const float* xtest  = (const float*)d_in[1];   // (4,75,3,84,84)
    const float* ytrain = (const float*)d_in[2];   // (4,25,5)
    // d_in[3] = ytest (unused)
    const float* w1 = (const float*)d_in[4];  const float* b1 = (const float*)d_in[5];
    const float* w2 = (const float*)d_in[6];  const float* b2 = (const float*)d_in[7];
    const float* w3 = (const float*)d_in[8];  const float* b3 = (const float*)d_in[9];
    const float* w4 = (const float*)d_in[10]; const float* b4 = (const float*)d_in[11];
    const float* wc = (const float*)d_in[12]; const float* bc = (const float*)d_in[13];
    float* out = (float*)d_out;

    // ---- workspace layout ----
    char* ws = (char*)d_ws;
    size_t off = 0;
    const size_t XB_ELEMS  = (size_t)400 * 3 * 84 * 84;          // bf16 images
    const size_t WB_ELEMS  = 1728 + 36864 + 73728 + 589824;      // bf16 weights
    const size_t A0_ELEMS  = (size_t)400 * 64 * 84 * 84;         // biggest conv out
    const size_t A1_ELEMS  = (size_t)400 * 64 * 42 * 42;         // biggest pool out

    __bf16* xb = (__bf16*)(ws + off); off = align256(off + XB_ELEMS * 2);
    __bf16* wb = (__bf16*)(ws + off); off = align256(off + WB_ELEMS * 2);
    __bf16* A0 = (__bf16*)(ws + off); off = align256(off + A0_ELEMS * 2);
    __bf16* A1 = (__bf16*)(ws + off); off = align256(off + A1_ELEMS * 2);
    float*  proto = (float*)(ws + off); off = align256(off + 4 * 5 * 512 * 4);

    __bf16* wb1 = wb;
    __bf16* wb2 = wb + 1728;
    __bf16* wb3 = wb + 1728 + 36864;
    __bf16* wb4 = wb + 1728 + 36864 + 73728;

    // ---- bf16 conversions ----
    {
        int n0 = 100 * 3 * 84 * 84, n1 = 300 * 3 * 84 * 84;
        f32_to_bf16_kernel<<<cdiv(n0, 256), 256, 0, stream>>>(xtrain, xb, n0);
        f32_to_bf16_kernel<<<cdiv(n1, 256), 256, 0, stream>>>(xtest, xb + n0, n1);
        f32_to_bf16_kernel<<<cdiv(1728, 256),   256, 0, stream>>>(w1, wb1, 1728);
        f32_to_bf16_kernel<<<cdiv(36864, 256),  256, 0, stream>>>(w2, wb2, 36864);
        f32_to_bf16_kernel<<<cdiv(73728, 256),  256, 0, stream>>>(w3, wb3, 73728);
        f32_to_bf16_kernel<<<cdiv(589824, 256), 256, 0, stream>>>(w4, wb4, 589824);
    }

    const int N = 400;
    // ---- block 1: 3 -> 64 @ 84, pool -> 42 ----
    conv_gemm_wmma<<<dim3(cdiv(N * 84 * 84, 128), cdiv(64, 64)), 256, 0, stream>>>(
        xb, wb1, b1, A0, N, 3, 84, 64);
    maxpool2_kernel<<<2048, 256, 0, stream>>>(A0, A1, N * 64, 84);
    // ---- block 2: 64 -> 64 @ 42, pool -> 21 ----
    conv_gemm_wmma<<<dim3(cdiv(N * 42 * 42, 128), cdiv(64, 64)), 256, 0, stream>>>(
        A1, wb2, b2, A0, N, 64, 42, 64);
    maxpool2_kernel<<<2048, 256, 0, stream>>>(A0, A1, N * 64, 42);
    // ---- block 3: 64 -> 128 @ 21, pool -> 10 ----
    conv_gemm_wmma<<<dim3(cdiv(N * 21 * 21, 128), cdiv(128, 64)), 256, 0, stream>>>(
        A1, wb3, b3, A0, N, 64, 21, 128);
    maxpool2_kernel<<<1024, 256, 0, stream>>>(A0, A1, N * 128, 21);
    // ---- block 4: 128 -> 512 @ 10, pool -> 5 ----
    conv_gemm_wmma<<<dim3(cdiv(N * 10 * 10, 128), cdiv(512, 64)), 256, 0, stream>>>(
        A1, wb4, b4, A0, N, 128, 10, 512);
    maxpool2_kernel<<<1024, 256, 0, stream>>>(A0, A1, N * 512, 10);
    // A1 now holds final features (400, 512, 5, 5) bf16

    // ---- heads ----
    proto_kernel<<<20, 256, 0, stream>>>(A1, ytrain, proto);
    classifier_kernel<<<300, 256, 0, stream>>>(A1, wc, bc, out);                   // 19200
    cosine_kernel<<<cdiv(300 * 25, 8), 256, 0, stream>>>(A1, proto, out + 19200);  // 37500
}
